// Mamba2Block_18184891532012
// MI455X (gfx1250) — matrix operand order repaired and checked by reading, not compiled, not run on hardware
//
#include <hip/hip_runtime.h>
#include <hip/hip_bf16.h>
#include <stdint.h>

// ---------------------------------------------------------------------------
// Mamba block for MI455X (gfx1250, wave32, WMMA).
// GEMMs run in bf16 via v_wmma_f32_16x16x32_bf16 with f32 accumulation.
// Big GEMMs stage weight tiles in LDS via GLOBAL_LOAD_ASYNC_TO_LDS_B128
// (double-buffered, ASYNCcnt-synchronized) so the DMA overlaps the WMMAs.
// ---------------------------------------------------------------------------

#define D_MODEL   1024
#define D_STATE   16
#define D_CONV    4
#define D_INNER   2048
#define DT_RANK   64
#define BB        2
#define LL        1024
#define MROWS     (BB * LL)      // 2048 rows for every GEMM
#define EPSV      1e-5f

typedef __attribute__((ext_vector_type(16))) __bf16 v16bf;
typedef __attribute__((ext_vector_type(8)))  float  v8f;
typedef __attribute__((ext_vector_type(4)))  int    v4i;

#if defined(__HIP_DEVICE_COMPILE__) && \
    __has_builtin(__builtin_amdgcn_global_load_async_to_lds_b128) && \
    __has_builtin(__builtin_amdgcn_s_wait_asynccnt)
#define USE_ASYNC_LDS 1
#else
#define USE_ASYNC_LDS 0
#endif

__device__ __forceinline__ unsigned short f32_to_bf16(float f) {
    unsigned int u = __float_as_uint(f);
    unsigned int r = (u + 0x7FFFu + ((u >> 16) & 1u)) >> 16;   // RNE
    return (unsigned short)r;
}

// ---------------------------------------------------------------------------
// fp32 -> bf16 bulk convert
// ---------------------------------------------------------------------------
__global__ __launch_bounds__(256)
void cvt_f32_bf16_kernel(const float* __restrict__ in,
                         unsigned short* __restrict__ out, int n) {
    int i = blockIdx.x * 256 + threadIdx.x;
    if (i < n) out[i] = f32_to_bf16(in[i]);
}

// ---------------------------------------------------------------------------
// hs = rmsnorm(hidden + residual) * w   -> bf16, one block per row
// ---------------------------------------------------------------------------
__global__ __launch_bounds__(256)
void rmsnorm_kernel(const float* __restrict__ hidden,
                    const float* __restrict__ residual,
                    const float* __restrict__ w,
                    unsigned short* __restrict__ out_bf16) {
    __shared__ float red[256];
    const int row = blockIdx.x;
    const int tid = threadIdx.x;
    float vals[D_MODEL / 256];
    float ss = 0.f;
#pragma unroll
    for (int j = 0; j < D_MODEL / 256; ++j) {
        int c = tid + j * 256;
        float v = hidden[(size_t)row * D_MODEL + c] + residual[(size_t)row * D_MODEL + c];
        vals[j] = v;
        ss += v * v;
    }
    red[tid] = ss;
    __syncthreads();
    for (int s = 128; s > 0; s >>= 1) {
        if (tid < s) red[tid] += red[tid + s];
        __syncthreads();
    }
    float rinv = rsqrtf(red[0] / (float)D_MODEL + EPSV);
#pragma unroll
    for (int j = 0; j < D_MODEL / 256; ++j) {
        int c = tid + j * 256;
        out_bf16[(size_t)row * D_MODEL + c] = f32_to_bf16(w[c] * vals[j] * rinv);
    }
}

// ---------------------------------------------------------------------------
// WMMA fragment layouts (wave32), as used here (contiguous K, same order for A and B):
//  A (16x32 bf16):  lane(half=lane>>4, r=lane&15) holds row M=r,
//                   K = k0+half*16..+15, contiguous  (2 x b128)
//  B (32x16 bf16):  lane holds column N=r, 16 contiguous K from k0+half*16
//  C (16x16 f32):   vgpr rr -> row rr+half*8, col r
// ---------------------------------------------------------------------------
union frag16 { v16bf v; uint4 u[2]; };

// ---------------------------------------------------------------------------
// LDS-staged WMMA GEMM: C[M,N] = A[M,K] * W[N,K]^T
// Block = 8 waves stacked along M: 128(M) x 64(N) tile per block.
// The 64x32 bf16 weight tile per K-step is staged once per block into a
// double-buffered LDS tile (async-to-LDS), then consumed by all 8 waves
// via ds_load_b128. All 8 ds_loads of a K-step are issued before the first
// WMMA so the dscnt waits are partial. N%64==0, K%32==0, M%128==0.
// ---------------------------------------------------------------------------
__global__ __launch_bounds__(256)
void wmma_gemm_lds_kernel(const unsigned short* __restrict__ A,
                          const unsigned short* __restrict__ W,
                          float* __restrict__ C,
                          int M, int N, int K) {
    __shared__ unsigned short Btile[2][64 * 32];   // 2 x 4KB

    const int lane = threadIdx.x & 31;
    const int wave = threadIdx.x >> 5;
    const int ngrp = N / 64;
    const int bm   = blockIdx.x / ngrp;
    const int bn   = blockIdx.x % ngrp;
    const int n0   = bn * 64;
    const int half = lane >> 4;
    const int r    = lane & 15;
    const int nsteps = K / 32;

    // cooperative stage of one 64x32 weight tile: 256 threads x one b128
    const int srow = threadIdx.x >> 2;        // 0..63
    const int skc  = (threadIdx.x & 3) * 8;   // 0,8,16,24 (bf16 elems)
    auto stage = [&](int ks, int buf) {
        const unsigned short* src = W + (size_t)(n0 + srow) * K + ks * 32 + skc;
        unsigned short* dst = &Btile[buf][srow * 32 + skc];
#if USE_ASYNC_LDS
        __builtin_amdgcn_global_load_async_to_lds_b128(
            (__attribute__((address_space(1))) v4i*)(unsigned short*)src,
            (__attribute__((address_space(3))) v4i*)dst, 0, 0);
#else
        *(uint4*)dst = *(const uint4*)src;
#endif
    };

    v8f acc[4];
#pragma unroll
    for (int i = 0; i < 4; ++i) acc[i] = (v8f){0.f,0.f,0.f,0.f,0.f,0.f,0.f,0.f};

    const unsigned short* Arow = A + (size_t)(bm * 128 + wave * 16 + r) * K;

    stage(0, 0);
    for (int ks = 0; ks < nsteps; ++ks) {
        const int cur = ks & 1;
#if USE_ASYNC_LDS
        __builtin_amdgcn_s_wait_asynccnt(0);
#endif
        __syncthreads();
        if (ks + 1 < nsteps) stage(ks + 1, cur ^ 1);

        // A fragment: 2 x global b128 (per-wave unique rows)
        frag16 af;
        af.u[0] = *(const uint4*)(Arow + ks * 32 + half * 16);
        af.u[1] = *(const uint4*)(Arow + ks * 32 + half * 16 + 8);

        // issue all 8 ds_load_b128 for the 4 B fragments first
        frag16 bfr[4];
#pragma unroll
        for (int i = 0; i < 4; ++i) {
            const unsigned short* Bp = &Btile[cur][(i * 16 + r) * 32 + half * 16];
            bfr[i].u[0] = *(const uint4*)(Bp);
            bfr[i].u[1] = *(const uint4*)(Bp + 8);
        }
        // then the 4 WMMAs (waits become partial, latency overlaps the pipe)
#pragma unroll
        for (int i = 0; i < 4; ++i)
            acc[i] = __builtin_amdgcn_wmma_f32_16x16x32_bf16(
                false, af.v, false, bfr[i].v, (short)0, acc[i], false, false);

        __syncthreads();   // all waves done reading Btile[cur] before restage
    }

#pragma unroll
    for (int i = 0; i < 4; ++i) {
        const int col = n0 + i * 16 + r;
#pragma unroll
        for (int rr = 0; rr < 8; ++rr) {
            const int row = bm * 128 + wave * 16 + rr + half * 8;
            C[(size_t)row * N + col] = acc[i][rr];
        }
    }
}

// ---------------------------------------------------------------------------
// Per-wave WMMA GEMM (for small / odd N, e.g. N=96): 16 x (16*NT) per wave.
// ---------------------------------------------------------------------------
template <int NT>
__global__ __launch_bounds__(256)
void wmma_gemm_bf16_kernel(const unsigned short* __restrict__ A,
                           const unsigned short* __restrict__ W,
                           float* __restrict__ C,
                           int M, int N, int K) {
    const int lane = threadIdx.x & 31;
    const int wave = threadIdx.x >> 5;
    const int ngrp = N / (16 * NT);
    const int gid  = blockIdx.x * 8 + wave;
    const int mt   = gid / ngrp;
    const int ng   = gid % ngrp;
    if (mt * 16 >= M) return;                  // uniform per wave: EXEC stays all-ones

    const int half = lane >> 4;
    const int r    = lane & 15;

    v8f acc[NT];
#pragma unroll
    for (int i = 0; i < NT; ++i) acc[i] = (v8f){0.f,0.f,0.f,0.f,0.f,0.f,0.f,0.f};

    const unsigned short* Arow = A + (size_t)(mt * 16 + r) * K;

    for (int k0 = 0; k0 < K; k0 += 32) {
        frag16 af;
        af.u[0] = *(const uint4*)(Arow + k0 + half * 16);
        af.u[1] = *(const uint4*)(Arow + k0 + half * 16 + 8);
        frag16 bfr[NT];
#pragma unroll
        for (int i = 0; i < NT; ++i) {
            const int n = (ng * NT + i) * 16 + r;
            const unsigned short* Wp = W + (size_t)n * K + k0 + half * 16;
            bfr[i].u[0] = *(const uint4*)(Wp);
            bfr[i].u[1] = *(const uint4*)(Wp + 8);
        }
#pragma unroll
        for (int i = 0; i < NT; ++i)
            acc[i] = __builtin_amdgcn_wmma_f32_16x16x32_bf16(
                false, af.v, false, bfr[i].v, (short)0, acc[i], false, false);
    }
#pragma unroll
    for (int i = 0; i < NT; ++i) {
        const int col = (ng * NT + i) * 16 + r;
#pragma unroll
        for (int rr = 0; rr < 8; ++rr) {
            const int row = mt * 16 + rr + half * 8;
            C[(size_t)row * N + col] = acc[i][rr];
        }
    }
}

// ---------------------------------------------------------------------------
// depthwise causal conv (taps=4) + bias + SiLU over the x half of xz
// ---------------------------------------------------------------------------
__global__ __launch_bounds__(256)
void conv_silu_kernel(const float* __restrict__ xz,
                      const float* __restrict__ conv_w,
                      const float* __restrict__ conv_b,
                      float* __restrict__ x_f32,
                      unsigned short* __restrict__ x_bf16) {
    int idx = blockIdx.x * 256 + threadIdx.x;
    if (idx >= MROWS * D_INNER) return;
    const int c = idx % D_INNER;
    const int t = (idx / D_INNER) % LL;
    const int b = idx / (D_INNER * LL);
    float s = conv_b[c];
#pragma unroll
    for (int j = 0; j < D_CONV; ++j) {
        int tt = t - (D_CONV - 1) + j;
        if (tt >= 0)
            s += conv_w[c * D_CONV + j] *
                 xz[((size_t)(b * LL + tt)) * (2 * D_INNER) + c];
    }
    float y = s / (1.f + __expf(-s));   // SiLU
    x_f32[idx]  = y;
    x_bf16[idx] = f32_to_bf16(y);
}

// ---------------------------------------------------------------------------
// dt slice of x_dbl -> bf16 A-matrix for dt_proj
// ---------------------------------------------------------------------------
__global__ __launch_bounds__(256)
void dt_extract_kernel(const float* __restrict__ x_dbl,
                       unsigned short* __restrict__ dtA) {
    int i = blockIdx.x * 256 + threadIdx.x;
    if (i >= MROWS * DT_RANK) return;
    int row = i / DT_RANK, c = i % DT_RANK;
    dtA[i] = f32_to_bf16(x_dbl[(size_t)row * (DT_RANK + 2 * D_STATE) + c]);
}

// ---------------------------------------------------------------------------
// dt = softplus(dt_lin + bias), in place
// ---------------------------------------------------------------------------
__global__ __launch_bounds__(256)
void softplus_kernel(float* __restrict__ dt, const float* __restrict__ bias) {
    int i = blockIdx.x * 256 + threadIdx.x;
    if (i >= MROWS * D_INNER) return;
    float v = dt[i] + bias[i % D_INNER];
    dt[i] = (v > 20.f) ? v : log1pf(__expf(v));
}

// ---------------------------------------------------------------------------
// selective scan: one lane per (b, d) channel, 16-wide state in registers.
// All 32 lanes of a wave share (b,t) -> B/C loads are wave-uniform & cached.
// ---------------------------------------------------------------------------
__global__ __launch_bounds__(256)
void scan_kernel(const float* __restrict__ dt,      // [M, D_INNER]
                 const float* __restrict__ x_dbl,   // [M, 96]
                 const float* __restrict__ x_f32,   // [M, D_INNER]
                 const float* __restrict__ A_log,   // [D_INNER, 16]
                 float* __restrict__ yscan) {       // [M, D_INNER]
    int tid = blockIdx.x * 256 + threadIdx.x;
    if (tid >= BB * D_INNER) return;
    const int d = tid % D_INNER;
    const int b = tid / D_INNER;

    float A[D_STATE], h[D_STATE];
#pragma unroll
    for (int s = 0; s < D_STATE; ++s) {
        A[s] = -__expf(A_log[d * D_STATE + s]);
        h[s] = 0.f;
    }
    for (int t = 0; t < LL; ++t) {
        const size_t row = (size_t)b * LL + t;
        const float dt_v = dt[row * D_INNER + d];
        const float x_v  = x_f32[row * D_INNER + d];
        const float dx   = dt_v * x_v;
        const float* bc  = x_dbl + row * (DT_RANK + 2 * D_STATE);
        float y = 0.f;
#pragma unroll
        for (int s = 0; s < D_STATE; ++s) {
            float dA = __expf(dt_v * A[s]);
            h[s] = dA * h[s] + dx * bc[DT_RANK + s];
            y   += h[s] * bc[DT_RANK + D_STATE + s];
        }
        yscan[row * D_INNER + d] = y;
    }
}

// ---------------------------------------------------------------------------
// y = (yscan + x*D) * silu(z)  -> bf16 A-matrix for out_proj
// ---------------------------------------------------------------------------
__global__ __launch_bounds__(256)
void gate_kernel(const float* __restrict__ yscan,
                 const float* __restrict__ x_f32,
                 const float* __restrict__ xz,
                 const float* __restrict__ Dv,
                 unsigned short* __restrict__ y_bf16) {
    int i = blockIdx.x * 256 + threadIdx.x;
    if (i >= MROWS * D_INNER) return;
    const int c = i % D_INNER;
    const int row = i / D_INNER;
    float z = xz[(size_t)row * (2 * D_INNER) + D_INNER + c];
    float g = z / (1.f + __expf(-z));
    float y = (yscan[i] + x_f32[i] * Dv[c]) * g;
    y_bf16[i] = f32_to_bf16(y);
}

// ---------------------------------------------------------------------------
// host-side launcher
// ---------------------------------------------------------------------------
static inline size_t align256(size_t x) { return (x + 255) & ~(size_t)255; }

extern "C" void kernel_launch(void* const* d_in, const int* in_sizes, int n_in,
                              void* d_out, int out_size, void* d_ws, size_t ws_size,
                              hipStream_t stream) {
    const float* hidden   = (const float*)d_in[0];
    const float* residual = (const float*)d_in[1];
    const float* norm_w   = (const float*)d_in[2];
    const float* in_proj  = (const float*)d_in[3];   // [4096, 1024]
    const float* conv_w   = (const float*)d_in[4];   // [2048, 1, 4]
    const float* conv_b   = (const float*)d_in[5];
    const float* x_proj   = (const float*)d_in[6];   // [96, 2048]
    const float* dt_proj  = (const float*)d_in[7];   // [2048, 64]
    const float* dt_b     = (const float*)d_in[8];
    const float* A_log    = (const float*)d_in[9];   // [2048, 16]
    const float* Dv       = (const float*)d_in[10];
    const float* out_proj = (const float*)d_in[11];  // [1024, 2048]
    float* out = (float*)d_out;

    // workspace carve-out
    char* p = (char*)d_ws;
    size_t off = 0;
    auto take = [&](size_t bytes) { char* q = p + off; off = align256(off + bytes); return (void*)q; };

    unsigned short* hs_bf   = (unsigned short*)take((size_t)MROWS * D_MODEL * 2);
    unsigned short* w_in    = (unsigned short*)take((size_t)4096 * 1024 * 2);
    unsigned short* w_xp    = (unsigned short*)take((size_t)96 * 2048 * 2);
    unsigned short* w_dt    = (unsigned short*)take((size_t)2048 * 64 * 2);
    unsigned short* w_out   = (unsigned short*)take((size_t)1024 * 2048 * 2);
    float*          xz      = (float*)take((size_t)MROWS * 4096 * 4);
    float*          x_f32   = (float*)take((size_t)MROWS * D_INNER * 4);
    unsigned short* x_bf    = (unsigned short*)take((size_t)MROWS * D_INNER * 2);
    float*          x_dbl   = (float*)take((size_t)MROWS * 96 * 4);
    unsigned short* dtA     = (unsigned short*)take((size_t)MROWS * DT_RANK * 2);
    float*          dt_f    = (float*)take((size_t)MROWS * D_INNER * 4);
    float*          yscan   = (float*)take((size_t)MROWS * D_INNER * 4);
    unsigned short* y_bf    = (unsigned short*)take((size_t)MROWS * D_INNER * 2);
    (void)ws_size; (void)n_in; (void)in_sizes; (void)out_size;

    const int T = 256;
    auto blocks = [](size_t n) { return (unsigned)((n + 255) / 256); };

    // weight downcasts
    cvt_f32_bf16_kernel<<<blocks((size_t)4096 * 1024), T, 0, stream>>>(in_proj,  w_in,  4096 * 1024);
    cvt_f32_bf16_kernel<<<blocks((size_t)96 * 2048),   T, 0, stream>>>(x_proj,   w_xp,  96 * 2048);
    cvt_f32_bf16_kernel<<<blocks((size_t)2048 * 64),   T, 0, stream>>>(dt_proj,  w_dt,  2048 * 64);
    cvt_f32_bf16_kernel<<<blocks((size_t)1024 * 2048), T, 0, stream>>>(out_proj, w_out, 1024 * 2048);

    // 1) hs = rmsnorm(hidden + residual) -> bf16
    rmsnorm_kernel<<<MROWS, T, 0, stream>>>(hidden, residual, norm_w, hs_bf);

    // 2) xz = hs @ in_proj^T     [2048, 4096], K=1024  (LDS-staged)
    wmma_gemm_lds_kernel<<<(MROWS / 128) * (4096 / 64), T, 0, stream>>>(
        hs_bf, w_in, xz, MROWS, 4096, 1024);

    // 3) depthwise conv + SiLU -> x
    conv_silu_kernel<<<blocks((size_t)MROWS * D_INNER), T, 0, stream>>>(xz, conv_w, conv_b, x_f32, x_bf);

    // 4) x_dbl = x @ x_proj^T    [2048, 96], K=2048  (per-wave, NT=2: 96 = 3*32)
    {
        unsigned waves = (MROWS / 16) * (96 / 32);
        wmma_gemm_bf16_kernel<2><<<(waves + 7) / 8, T, 0, stream>>>(x_bf, w_xp, x_dbl, MROWS, 96, 2048);
    }

    // 5) dt slice -> bf16, dt_lin = dtA @ dt_proj^T   [2048, 2048], K=64  (LDS-staged)
    dt_extract_kernel<<<blocks((size_t)MROWS * DT_RANK), T, 0, stream>>>(x_dbl, dtA);
    wmma_gemm_lds_kernel<<<(MROWS / 128) * (D_INNER / 64), T, 0, stream>>>(
        dtA, w_dt, dt_f, MROWS, D_INNER, DT_RANK);
    softplus_kernel<<<blocks((size_t)MROWS * D_INNER), T, 0, stream>>>(dt_f, dt_b);

    // 6) selective scan
    scan_kernel<<<(BB * D_INNER) / T, T, 0, stream>>>(dt_f, x_dbl, x_f32, A_log, yscan);

    // 7) gate -> bf16
    gate_kernel<<<blocks((size_t)MROWS * D_INNER), T, 0, stream>>>(yscan, x_f32, xz, Dv, y_bf);

    // 8) out = y @ out_proj^T    [2048, 1024], K=2048  (LDS-staged) -> d_out
    wmma_gemm_lds_kernel<<<(MROWS / 128) * (1024 / 64), T, 0, stream>>>(
        y_bf, w_out, out, MROWS, 1024, D_INNER);

    // 9) second tuple output: residual passthrough
    (void)hipMemcpyAsync(out + (size_t)MROWS * D_MODEL, residual,
                         (size_t)MROWS * D_MODEL * sizeof(float),
                         hipMemcpyDeviceToDevice, stream);
}
